// SpeBranch_72533407694912
// MI455X (gfx1250) — compile-verified, matrix operand order checked by reading, not run on hardware
//
#include <hip/hip_runtime.h>
#include <stdint.h>

// ---------------------------------------------------------------------------
// Problem constants (reference: N=2, B=128, H=32, W=32, C=128, G=8, shift=4)
// ---------------------------------------------------------------------------
#define NN_   2
#define BB_   128
#define HH_   32
#define WW_   32
#define CC_   128
#define C1_   64
#define GG_   8
#define BG_   16      // BB_/GG_
#define SHIFT_ 4
#define HW_   1024    // HH_*WW_
#define M0_   262144  // NN_*BB_*HW_

typedef __attribute__((ext_vector_type(16))) __bf16 v16bf;
typedef __attribute__((ext_vector_type(8)))  float  v8f;

#define DEVINL __device__ __forceinline__

DEVINL unsigned short f2bf(float x) {
  union { float f; uint32_t u; } v; v.f = x;
  uint32_t r = v.u + 0x7FFFu + ((v.u >> 16) & 1u);   // round-to-nearest-even
  return (unsigned short)(r >> 16);
}

DEVINL float gelu_exact(float x) {
  return 0.5f * x * (1.0f + erff(x * 0.7071067811865475f));
}

union Frag16 { v16bf v; unsigned short u[16]; };

// Load a 16x32 bf16 fragment (row-major, row stride 32 elems) per ISA layout:
// lanes 0-15: row=lane, K in {0..7}U{16..23}; lanes 16-31: row=lane-16,
// K in {8..15}U{24..31}; packed 2 per VGPR.
DEVINL v16bf ld_frag(const unsigned short* base, int lane) {
  int row = lane & 15;
  int ko  = (lane & 16) ? 8 : 0;
  const unsigned short* p = base + row * 32 + ko;
  Frag16 f;
#pragma unroll
  for (int i = 0; i < 8; ++i) { f.u[i] = p[i]; f.u[i + 8] = p[i + 16]; }
  return f.v;
}

// Cooperative ROWS x 32 bf16 tile fetch, global -> LDS via CDNA5 async path
// (no VGPR round-trip; tracked by ASYNCcnt). Each thread issues ELEMS/8
// GLOBAL_LOAD_ASYNC_TO_LDS_B128 ops.
template<int ROWS>
DEVINL void tile_async(const unsigned short* g, long ldg, unsigned short* s, int tid) {
  constexpr int ELEMS = ROWS * 32 / 256;  // elems per thread (16 or 8)
  constexpr int TPR   = 32 / ELEMS;       // threads per row (2 or 4)
  int r   = tid / TPR;
  int off = (tid % TPR) * ELEMS;
  const unsigned short* gp = g + (long)r * ldg + off;
  unsigned lp = (unsigned)(uintptr_t)(s + r * 32 + off);  // LDS byte offset
#pragma unroll
  for (int i = 0; i < ELEMS / 8; ++i) {
    unsigned long long ga = (unsigned long long)(uintptr_t)gp + (unsigned)(i * 16);
    asm volatile("global_load_async_to_lds_b128 %0, %1, off"
                 :: "v"(lp + (unsigned)(i * 16)), "v"(ga) : "memory");
  }
}

// ---------------------------------------------------------------------------
// Generic bf16 GEMM: D[M x Ntot] = A[M x K] * Wt^T + bias, Wt stored [Ntot][K].
// Block tile 128 x BN, K step 32, 8 wave32s in 4(M) x 2(N) grid; each wave
// owns a 32 x (BN/2) sub-tile = 2 x (BN/32) wmma accumulators.
// Double-buffered LDS, async global->LDS copies overlapping WMMA compute,
// final K iteration peeled so the steady-state loop is branch-free.
// ---------------------------------------------------------------------------
template<int AN>
DEVINL void wmma_tile(const unsigned short* sA, const unsigned short* sB,
                      int wm, int wn, int lane, v8f (&acc)[2][AN]) {
  constexpr int BK = 32;
  constexpr int WN = AN * 16;
  v16bf af[2];
#pragma unroll
  for (int i = 0; i < 2; ++i)
    af[i] = ld_frag(&sA[(wm * 32 + i * 16) * BK], lane);
#pragma unroll
  for (int j = 0; j < AN; ++j) {
    v16bf bfrag = ld_frag(&sB[(wn * WN + j * 16) * BK], lane);
#pragma unroll
    for (int i = 0; i < 2; ++i)
      acc[i][j] = __builtin_amdgcn_wmma_f32_16x16x32_bf16(
          false, af[i], false, bfrag, (short)0, acc[i][j], false, false);
  }
}

template<int BN, bool OUT_BF16, bool DO_GELU>
__global__ __launch_bounds__(256, 2)
void gemm_bf16_kernel(const unsigned short* __restrict__ A, long lda,
                      const unsigned short* __restrict__ Wt,
                      const float* __restrict__ bias,
                      void* __restrict__ Dv, long ldd, int K)
{
  constexpr int BM = 128, BK = 32;
  constexpr int WN = BN / 2;   // wave N span
  constexpr int AN = WN / 16;  // accumulators along N per wave
  // per-wave async ops per (A tile + B tile): A = 2, B = BN/64
  constexpr int OPS = (BM * 32 / 256 / 8) + (BN * 32 / 256 / 8);
  __shared__ unsigned short sA[2][BM * BK];
  __shared__ unsigned short sB[2][BN * BK];

  const int tid  = threadIdx.x;
  const int lane = tid & 31;
  const int wave = tid >> 5;
  const int wm   = wave & 3;   // 0..3 -> 32-row slab
  const int wn   = wave >> 2;  // 0..1 -> WN-col slab
  const long m0 = (long)blockIdx.x * BM;
  const long n0 = (long)blockIdx.y * BN;

  v8f acc[2][AN];
#pragma unroll
  for (int i = 0; i < 2; ++i)
#pragma unroll
    for (int j = 0; j < AN; ++j)
#pragma unroll
      for (int r = 0; r < 8; ++r) acc[i][j][r] = 0.0f;

  // prologue: fetch tile 0 into buffer 0
  tile_async<BM>(A  + m0 * lda,     lda,     sA[0], tid);
  tile_async<BN>(Wt + n0 * (long)K, (long)K, sB[0], tid);

  const int kIter = K / BK;
  // steady state: branch-free (last iteration peeled)
  for (int t = 0; t < kIter - 1; ++t) {
    const int cur = t & 1;
    const long kn = (long)(t + 1) * BK;
    tile_async<BM>(A  + m0 * lda     + kn, lda,     sA[cur ^ 1], tid);
    tile_async<BN>(Wt + n0 * (long)K + kn, (long)K, sB[cur ^ 1], tid);
    // wait until only the freshly issued group is outstanding
    asm volatile("s_wait_asynccnt %0" :: "n"(OPS) : "memory");
    __syncthreads();   // all waves' portions of the current tile are in LDS
    wmma_tile<AN>(sA[cur], sB[cur], wm, wn, lane, acc);
    __syncthreads();   // protect buffer reused by next iteration's async fill
  }
  // tail: last tile
  asm volatile("s_wait_asynccnt 0x0" ::: "memory");
  __syncthreads();
  wmma_tile<AN>(sA[(kIter - 1) & 1], sB[(kIter - 1) & 1], wm, wn, lane, acc);

  // C/D layout: lane<16 -> N=lane, M=vgpr; lane>=16 -> N=lane-16, M=8+vgpr
  const int cm = (lane >> 4) << 3;
  const int cn = lane & 15;
#pragma unroll
  for (int i = 0; i < 2; ++i)
#pragma unroll
    for (int j = 0; j < AN; ++j) {
      long nn = n0 + wn * WN + j * 16 + cn;
      float b = bias[nn];
#pragma unroll
      for (int r = 0; r < 8; ++r) {
        long mm = m0 + wm * 32 + i * 16 + cm + r;
        float v = acc[i][j][r] + b;
        if (DO_GELU) v = gelu_exact(v);
        if (OUT_BF16) ((unsigned short*)Dv)[mm * ldd + nn] = f2bf(v);
        else          ((float*)Dv)[mm * ldd + nn] = v;
      }
    }
}

// ---------------------------------------------------------------------------
// Conversion / pack kernels (bandwidth-only permutations)
// ---------------------------------------------------------------------------
__global__ void f32_to_bf16_kernel(const float* __restrict__ s,
                                   unsigned short* __restrict__ d, long n) {
  long i = (long)blockIdx.x * blockDim.x + threadIdx.x;
  if (i < n) d[i] = f2bf(s[i]);
}

// w[K][N] f32  ->  wt[N][K] bf16
__global__ void wtrans_kernel(const float* __restrict__ w,
                              unsigned short* __restrict__ wt, int K, int N) {
  long i = (long)blockIdx.x * blockDim.x + threadIdx.x;
  if (i >= (long)K * N) return;
  int k = (int)(i / N), n = (int)(i % N);
  wt[(long)n * K + k] = f2bf(w[i]);
}

// A[row=(n,g,h,w)][c*BG+b] = Y[n, (g*BG+b+shift)%B, h, w, 64+c]
__global__ void pack_band_kernel(const unsigned short* __restrict__ Y,
                                 unsigned short* __restrict__ A, int shift) {
  long i = (long)blockIdx.x * blockDim.x + threadIdx.x;   // < 16384*1024
  if (i >= 16384L * 1024) return;
  int f = (int)(i & 1023); long row = i >> 10;
  int c = f >> 4, b = f & 15;
  int hw = (int)(row & 1023); int rg = (int)(row >> 10);
  int g = rg & 7, n = rg >> 3;
  int band = (g * BG_ + b + shift) & (BB_ - 1);
  A[i] = Y[(((long)(n * BB_ + band) * HW_ + hw)) * CC_ + C1_ + c];
}

// XC[pix][j]: j<64 -> x21 channel j; j>=64 -> rolled-back x22 channel j-64
__global__ void pack_xc_kernel(const unsigned short* __restrict__ T21,
                               const unsigned short* __restrict__ T22,
                               unsigned short* __restrict__ XC) {
  long i = (long)blockIdx.x * blockDim.x + threadIdx.x;   // < M0*128
  if (i >= (long)M0_ * CC_) return;
  int j = (int)(i & 127); long pix = i >> 7;
  int hw = (int)(pix & 1023); int nb = (int)(pix >> 10);
  int band = nb & (BB_ - 1); int n = nb >> 7;
  unsigned short v;
  if (j < C1_) {
    int g = band >> 4, b = band & 15;
    long r = (long)(n * GG_ + g) * HW_ + hw;
    v = T21[r * 1024 + j * BG_ + b];
  } else {
    int c  = j - C1_;
    int bb = (band - SHIFT_) & (BB_ - 1);
    int g = bb >> 4, b = bb & 15;
    long r = (long)(n * GG_ + g) * HW_ + hw;
    v = T22[r * 1024 + c * BG_ + b];
  }
  XC[i] = v;
}

// A41[row=(n,b,h,w)][c*G+g] = XCout[(n, g*BG+b, h, w)][c]
__global__ void pack_a41_kernel(const unsigned short* __restrict__ XCout,
                                unsigned short* __restrict__ A41) {
  long i = (long)blockIdx.x * blockDim.x + threadIdx.x;   // < 32768*512
  if (i >= 32768L * 512) return;
  int f = (int)(i & 511); long row = i >> 9;
  int c = f >> 3, g = f & 7;
  int hw = (int)(row & 1023); int nb = (int)(row >> 10);
  int b = nb & 15, n = nb >> 4;
  int band = g * BG_ + b;
  A41[i] = XCout[((long)(n * BB_ + band) * HW_ + hw) * C1_ + c];
}

// F[pix][j]: j<64 -> x1out[pix][j]; j>=64 -> x4 channel (j-64) from T41
__global__ void pack_f_kernel(const unsigned short* __restrict__ x1out,
                              const unsigned short* __restrict__ T41,
                              unsigned short* __restrict__ F) {
  long i = (long)blockIdx.x * blockDim.x + threadIdx.x;   // < M0*128
  if (i >= (long)M0_ * CC_) return;
  int j = (int)(i & 127); long pix = i >> 7;
  unsigned short v;
  if (j < C1_) {
    v = x1out[pix * C1_ + j];
  } else {
    int hw = (int)(pix & 1023); int nb = (int)(pix >> 10);
    int band = nb & (BB_ - 1); int n = nb >> 7;
    int c = j - C1_;
    int g = band >> 4, b = band & 15;
    v = T41[((long)(n * BG_ + b) * HW_ + hw) * 512 + c * GG_ + g];
  }
  F[i] = v;
}

// ---------------------------------------------------------------------------
// Launch sequence
// ---------------------------------------------------------------------------
extern "C" void kernel_launch(void* const* d_in, const int* in_sizes, int n_in,
                              void* d_out, int out_size, void* d_ws, size_t ws_size,
                              hipStream_t stream) {
  (void)in_sizes; (void)n_in; (void)out_size; (void)ws_size;
  const float* x_f32     = (const float*)d_in[0];
  const float* head_w    = (const float*)d_in[1];
  const float* head_b    = (const float*)d_in[2];
  const float* mlp1_w    = (const float*)d_in[3];
  const float* mlp1_b    = (const float*)d_in[4];
  const float* mlp21_w   = (const float*)d_in[5];
  const float* mlp21_b   = (const float*)d_in[6];
  const float* mlp31_w   = (const float*)d_in[7];
  const float* mlp31_b   = (const float*)d_in[8];
  const float* mlp41_w   = (const float*)d_in[9];
  const float* mlp41_b   = (const float*)d_in[10];
  const float* concate_w = (const float*)d_in[11];
  const float* concate_b = (const float*)d_in[12];
  const float* fusion_w  = (const float*)d_in[13];
  const float* fusion_b  = (const float*)d_in[14];

  // --- workspace carve-up (bf16 buffers as ushort) ---
  char* ws = (char*)d_ws;
  size_t off = 0;
  auto carve = [&](size_t bytes) -> char* {
    char* p = ws + off;
    off = (off + bytes + 255) & ~(size_t)255;
    return p;
  };
  unsigned short* Wt_head = (unsigned short*)carve(128 * 128 * 2);
  unsigned short* Wt_mlp1 = (unsigned short*)carve(64 * 64 * 2);
  unsigned short* Wt_21   = (unsigned short*)carve(1024L * 1024 * 2);
  unsigned short* Wt_31   = (unsigned short*)carve(1024L * 1024 * 2);
  unsigned short* Wt_41   = (unsigned short*)carve(512 * 512 * 2);
  unsigned short* Wt_cat  = (unsigned short*)carve(64 * 128 * 2);
  unsigned short* Wt_fus  = (unsigned short*)carve(128 * 128 * 2);
  unsigned short* Xbf     = (unsigned short*)carve((long)M0_ * CC_ * 2); // reused: XC, then F
  unsigned short* Ybuf    = (unsigned short*)carve((long)M0_ * CC_ * 2); // reused: A41
  unsigned short* x1out   = (unsigned short*)carve((long)M0_ * C1_ * 2);
  unsigned short* P1      = (unsigned short*)carve(16384L * 1024 * 2);  // A21 -> A22 -> XCout
  unsigned short* P2      = (unsigned short*)carve(16384L * 1024 * 2);  // T21 -> T41
  unsigned short* P3      = (unsigned short*)carve(16384L * 1024 * 2);  // T22

  const long NX = (long)M0_ * CC_;   // 33.5M elems

  // --- weight transpose+convert, input convert ---
  wtrans_kernel<<<(128 * 128 + 255) / 256, 256, 0, stream>>>(head_w, Wt_head, 128, 128);
  wtrans_kernel<<<(64 * 64 + 255) / 256, 256, 0, stream>>>(mlp1_w, Wt_mlp1, 64, 64);
  wtrans_kernel<<<(1024 * 1024 + 255) / 256, 256, 0, stream>>>(mlp21_w, Wt_21, 1024, 1024);
  wtrans_kernel<<<(1024 * 1024 + 255) / 256, 256, 0, stream>>>(mlp31_w, Wt_31, 1024, 1024);
  wtrans_kernel<<<(512 * 512 + 255) / 256, 256, 0, stream>>>(mlp41_w, Wt_41, 512, 512);
  wtrans_kernel<<<(128 * 64 + 255) / 256, 256, 0, stream>>>(concate_w, Wt_cat, 128, 64);
  wtrans_kernel<<<(128 * 128 + 255) / 256, 256, 0, stream>>>(fusion_w, Wt_fus, 128, 128);
  f32_to_bf16_kernel<<<(int)(NX / 256), 256, 0, stream>>>(x_f32, Xbf, NX);

  // 1) head + GELU: Y = gelu(X @ head_w + b)           [M0 x 128]
  gemm_bf16_kernel<128, true, true><<<dim3(M0_ / 128, 1), 256, 0, stream>>>(
      Xbf, 128, Wt_head, head_b, Ybuf, 128, 128);

  // 2) x1 = Y[:, :64] @ mlp1_w + b                      [M0 x 64]
  gemm_bf16_kernel<64, true, false><<<dim3(M0_ / 128, 1), 256, 0, stream>>>(
      Ybuf, 128, Wt_mlp1, mlp1_b, x1out, 64, 64);

  // 3) band-mix branch 1: gather rows, GEMM with mlp21  [16384 x 1024]
  pack_band_kernel<<<(int)(16384L * 1024 / 256), 256, 0, stream>>>(Ybuf, P1, 0);
  gemm_bf16_kernel<128, true, false><<<dim3(16384 / 128, 1024 / 128), 256, 0, stream>>>(
      P1, 1024, Wt_21, mlp21_b, P2, 1024, 1024);

  // 4) band-mix branch 2 (shifted): GEMM with mlp31     [16384 x 1024]
  pack_band_kernel<<<(int)(16384L * 1024 / 256), 256, 0, stream>>>(Ybuf, P1, SHIFT_);
  gemm_bf16_kernel<128, true, false><<<dim3(16384 / 128, 1024 / 128), 256, 0, stream>>>(
      P1, 1024, Wt_31, mlp31_b, P3, 1024, 1024);

  // 5) concat channels (roll branch 2 back) -> XC, project to C1
  unsigned short* XC = Xbf;      // Xbf dead after head GEMM
  pack_xc_kernel<<<(int)(NX / 256), 256, 0, stream>>>(P2, P3, XC);
  unsigned short* XCout = P1;    // A22 dead after T22 GEMM
  gemm_bf16_kernel<64, true, false><<<dim3(M0_ / 128, 1), 256, 0, stream>>>(
      XC, 128, Wt_cat, concate_b, XCout, 64, 128);

  // 6) group mixing: gather (C*g) rows, GEMM with mlp41 [32768 x 512]
  unsigned short* A41 = Ybuf;    // Y dead
  pack_a41_kernel<<<(int)(32768L * 512 / 256), 256, 0, stream>>>(XCout, A41);
  unsigned short* T41 = P2;      // T21 dead after pack_xc
  gemm_bf16_kernel<128, true, false><<<dim3(32768 / 128, 512 / 128), 256, 0, stream>>>(
      A41, 512, Wt_41, mlp41_b, T41, 512, 512);

  // 7) fusion: concat(x1, x4) @ fusion_w + b -> f32 out [M0 x 128]
  unsigned short* F = Xbf;       // XC dead after concate GEMM
  pack_f_kernel<<<(int)(NX / 256), 256, 0, stream>>>(x1out, T41, F);
  gemm_bf16_kernel<128, false, false><<<dim3(M0_ / 128, 1), 256, 0, stream>>>(
      F, 128, Wt_fus, fusion_b, (float*)d_out, 128, 128);
}
